// MultiHeadedAttention_18322330485006
// MI455X (gfx1250) — matrix operand order repaired and checked by reading, not compiled, not run on hardware
//
#include <hip/hip_runtime.h>

// ---------------------------------------------------------------------------
// MHA forward for MI455X (gfx1250, wave32, WMMA bf16 16x16x32, f32 accum,
// TDM async bias/mask staging).  B=16 H=16 N=512 d_model=1024 d_k=64
// ---------------------------------------------------------------------------

typedef __bf16 bf16;
typedef __attribute__((ext_vector_type(16))) __bf16 v16bf;
typedef __attribute__((ext_vector_type(8)))  __bf16 v8bf;
typedef __attribute__((ext_vector_type(8)))  float  v8f;
typedef __attribute__((ext_vector_type(4)))  float  v4f;
typedef __attribute__((ext_vector_type(4)))  unsigned int u32x4;
typedef __attribute__((ext_vector_type(8)))  int  i32x8;
typedef __attribute__((ext_vector_type(4)))  int  i32x4;

#define D_MODEL 1024
#define NSEQ    512
#define NB      16
#define NH      16
#define DKH     64

// ---- WMMA wrapper: D = A(16x32 bf16) * B(32x16 bf16) + C(16x16 f32) -------
__device__ __forceinline__ v8f wmma_bf16(v16bf a, v16bf b, v8f c) {
  return __builtin_amdgcn_wmma_f32_16x16x32_bf16(
      /*neg_a=*/false, a, /*neg_b=*/false, b,
      /*c_mod=*/(short)0, c, /*reuse_a=*/false, /*reuse_b=*/false);
}

// ---- loaders matching the ISA wave32 VGPR layouts --------------------------
// B fragment: lane n holds column n, K = {kb..kb+7, 16+kb..16+kb+7}, kb=hi*8.
__device__ __forceinline__ v16bf ld16_bf(const bf16* p) {
  v8bf lo = *(const v8bf*)p;
  v8bf hi = *(const v8bf*)(p + 16);
  v16bf r;
#pragma unroll
  for (int e = 0; e < 8; ++e) { r[e] = lo[e]; r[e + 8] = hi[e]; }
  return r;
}
// A fragment: lane m holds row m, K = {kb..kb+7, 16+kb..16+kb+7}, kb=hi*8.
__device__ __forceinline__ v16bf ld8x2_gap_bf(const bf16* p) {
  v8bf lo = *(const v8bf*)p;
  v8bf hi = *(const v8bf*)(p + 16);
  v16bf r;
#pragma unroll
  for (int e = 0; e < 8; ++e) { r[e] = lo[e]; r[e + 8] = hi[e]; }
  return r;
}
// A fragment sourced from fp32 memory (converted in VALU).
__device__ __forceinline__ v16bf cvt8x2_gap_f32(const float* p) {
  v4f f0 = *(const v4f*)(p + 0),  f1 = *(const v4f*)(p + 4);
  v4f f2 = *(const v4f*)(p + 16), f3 = *(const v4f*)(p + 20);
  v16bf r;
#pragma unroll
  for (int e = 0; e < 4; ++e) {
    r[e]      = (bf16)f0[e];  r[e + 4]  = (bf16)f1[e];
    r[e + 8]  = (bf16)f2[e];  r[e + 12] = (bf16)f3[e];
  }
  return r;
}

// ---- Tensor Data Mover: 1-D DWORD copy global -> LDS (ISA 08 §8) ----------
// On AMDGCN the low 32 bits of a generic pointer to __shared__ are the
// workgroup-relative LDS byte offset (flat LDS-aperture mapping).
__device__ __forceinline__ unsigned lds_off_u32(const void* p) {
  return (unsigned)(unsigned long long)(uintptr_t)p;
}
__device__ __forceinline__ void tdm_copy_dwords_to_lds(const void* lds_dst,
                                                       const void* gsrc,
                                                       unsigned ndwords) {
  const unsigned long long ga = (unsigned long long)(uintptr_t)gsrc;
  u32x4 g0;
  g0[0] = 1u;                                   // count=1, user descriptor
  g0[1] = lds_off_u32(lds_dst);                 // lds_addr (bytes)
  g0[2] = (unsigned)ga;                         // global_addr[31:0]
  g0[3] = (unsigned)((ga >> 32) & 0x1FFFFFFu)   // global_addr[56:32]
        | (2u << 30);                           // type = 2 ("image")
  i32x8 g1;
  g1[0] = (int)(2u << 16);                      // data_size = 4 bytes
  g1[1] = (int)((ndwords & 0xFFFFu) << 16);     // tensor_dim0[15:0]
  g1[2] = (int)((ndwords >> 16) | (1u << 16));  // tensor_dim0[31:16], dim1=1
  g1[3] = (int)((ndwords & 0xFFFFu) << 16);     // tile_dim0 = ndwords (<=65535)
  g1[4] = 0;                                    // tile_dim1=0, tile_dim2=0
  g1[5] = (int)ndwords;                         // tensor_dim0_stride
  g1[6] = 0;
  g1[7] = 0;
  i32x4 gz = {0, 0, 0, 0};
#if __clang_major__ >= 23
  i32x8 gz8 = {0, 0, 0, 0, 0, 0, 0, 0};
  __builtin_amdgcn_tensor_load_to_lds(g0, g1, gz, gz, gz8, 0);
#else
  __builtin_amdgcn_tensor_load_to_lds(g0, g1, gz, gz, 0);
#endif
}

// ---------------------------------------------------------------------------
// Kernel 0: one-shot fp32 -> bf16 conversion of the 4 weight matrices into
// Wb[4][1024][1024].  512 blocks/matrix * 256 thr * 8 elem = 1M elems each.
// ---------------------------------------------------------------------------
__global__ __launch_bounds__(256) void cvt_w_kernel(
    const float* __restrict__ Wq, const float* __restrict__ Wk,
    const float* __restrict__ Wv, const float* __restrict__ Wo,
    bf16* __restrict__ Wb)
{
  const int which = blockIdx.x >> 9;
  const int idx   = ((blockIdx.x & 511) * 256 + threadIdx.x) * 8;
  const float* src = (which == 0) ? Wq : (which == 1) ? Wk : (which == 2) ? Wv : Wo;
  v4f f0 = *(const v4f*)(src + idx);
  v4f f1 = *(const v4f*)(src + idx + 4);
  v8bf r;
#pragma unroll
  for (int e = 0; e < 4; ++e) { r[e] = (bf16)f0[e]; r[e + 4] = (bf16)f1[e]; }
  *(v8bf*)(Wb + (size_t)which * D_MODEL * D_MODEL + idx) = r;
}

// ---------------------------------------------------------------------------
// Kernel 1: fused QKV projection, 2x2 register-blocked (wave owns 32x32 out).
// Outputs: Qh/Kh [bh][n][d] bf16, Vt [bh][d][n] bf16.
// ---------------------------------------------------------------------------
__global__ __launch_bounds__(256) void qkv_proj_kernel(
    const float* __restrict__ q, const float* __restrict__ k, const float* __restrict__ v,
    const bf16* __restrict__ Wb,
    const float* __restrict__ bq, const float* __restrict__ bk, const float* __restrict__ bv,
    bf16* __restrict__ Qh, bf16* __restrict__ Kh, bf16* __restrict__ Vt)
{
  const int wave = threadIdx.x >> 5;
  const int lane = threadIdx.x & 31;
  const int lm   = lane & 15;
  const int hi   = lane >> 4;

  const int WT_PER_MAT = (NB * NSEQ / 32) * (D_MODEL / 32);   // 8192
  const int tile  = blockIdx.x * 8 + wave;
  const int which = tile / WT_PER_MAT;
  const int rem   = tile - which * WT_PER_MAT;
  const int tm2   = rem >> 5;
  const int tn2   = rem & 31;

  const float* X  = (which == 0) ? q  : (which == 1) ? k  : v;
  const float* bs = (which == 0) ? bq : (which == 1) ? bk : bv;
  const bf16*  W  = Wb + (size_t)which * D_MODEL * D_MODEL;

  const int r0 = tm2 * 32, c0 = tn2 * 32;
  const float* xrow0 = X + (size_t)(r0 + lm) * D_MODEL;
  const float* xrow1 = X + (size_t)(r0 + 16 + lm) * D_MODEL;
  const bf16*  wrow0 = W + (size_t)(c0 + lm) * D_MODEL;
  const bf16*  wrow1 = W + (size_t)(c0 + 16 + lm) * D_MODEL;

  v8f acc[2][2] = {};
#pragma unroll 2
  for (int k0 = 0; k0 < D_MODEL; k0 += 32) {
    v16bf a0 = cvt8x2_gap_f32(xrow0 + k0 + hi * 8);
    v16bf a1 = cvt8x2_gap_f32(xrow1 + k0 + hi * 8);
    v16bf b0 = ld16_bf(wrow0 + k0 + hi * 8);
    v16bf b1 = ld16_bf(wrow1 + k0 + hi * 8);
    acc[0][0] = wmma_bf16(a0, b0, acc[0][0]);
    acc[0][1] = wmma_bf16(a0, b1, acc[0][1]);
    acc[1][0] = wmma_bf16(a1, b0, acc[1][0]);
    acc[1][1] = wmma_bf16(a1, b1, acc[1][1]);
  }

#pragma unroll
  for (int ni = 0; ni < 2; ++ni) {
    const int o  = c0 + ni * 16 + lm;
    const float bb = bs[o];
    const int h = o >> 6, d = o & 63;
#pragma unroll
    for (int mi = 0; mi < 2; ++mi) {
#pragma unroll
      for (int r = 0; r < 8; ++r) {
        const int g  = r0 + mi * 16 + r + hi * 8;
        const int gb = g >> 9, n = g & 511;
        const bf16 val = (bf16)(acc[mi][ni][r] + bb);
        if (which == 0)
          Qh[(((size_t)(gb * NH + h)) * NSEQ + n) * DKH + d] = val;
        else if (which == 1)
          Kh[(((size_t)(gb * NH + h)) * NSEQ + n) * DKH + d] = val;
        else
          Vt[(((size_t)(gb * NH + h)) * DKH + d) * NSEQ + n] = val;
      }
    }
  }
}

// ---------------------------------------------------------------------------
// Kernel 2: attention for one (b, h, 16-row tile).  4 waves / 128 threads.
//   TDM: bias tile (32 KB) and mask tile (32 KB) -> LDS, async under phase 1
//   S[16][512] = (Q_tile @ K^T)/8                           (WMMA -> LDS)
//   softmax with LDS bias add + mask                        (v_exp_f32)
//   ctx_tile[16][64] = P @ V                                (WMMA, A from LDS)
// ---------------------------------------------------------------------------
__global__ __launch_bounds__(128) void attn_kernel(
    const bf16* __restrict__ Qh, const bf16* __restrict__ Kh, const bf16* __restrict__ Vt,
    const float* __restrict__ abias, const int* __restrict__ mask,
    bf16* __restrict__ ctx)
{
  __shared__ float S [16][NSEQ];   // 32 KB raw scores / exp scratch
  __shared__ float Sb[16][NSEQ];   // 32 KB bias tile   (TDM)
  __shared__ int   Sm[16][NSEQ];   // 32 KB mask tile   (TDM)
  __shared__ bf16  P [16][NSEQ];   // 16 KB probabilities
  __shared__ float red[16][8];

  const int wave = threadIdx.x >> 5;
  const int lane = threadIdx.x & 31;
  const int lm   = lane & 15;
  const int hi   = lane >> 4;

  const int bh = blockIdx.x >> 5;          // b*16 + h
  const int it = blockIdx.x & 31;          // row tile
  const int b  = bh >> 4, h = bh & 15;
  const int i0 = it * 16;

  const bf16* Qb = Qh + (size_t)bh * NSEQ * DKH;
  const bf16* Kb = Kh + (size_t)bh * NSEQ * DKH;
  const bf16* Vb = Vt + (size_t)bh * DKH * NSEQ;

  // Kick off the async TDM copies of this block's bias & mask tiles (both
  // fully contiguous 16x512 dword regions) while WMMA computes raw scores.
  if (wave == 0) {
    tdm_copy_dwords_to_lds(&Sb[0][0],
                           abias + ((size_t)bh * NSEQ + i0) * NSEQ, 16 * NSEQ);
    tdm_copy_dwords_to_lds(&Sm[0][0],
                           mask + ((size_t)b * NSEQ + i0) * NSEQ, 16 * NSEQ);
  }
  // Warm L2 for the Vt slice consumed in phase 3 (global_prefetch_b8).
  {
    const char* vpf = (const char*)Vb;
#pragma unroll
    for (int t = 0; t < 4; ++t)
      __builtin_prefetch(vpf + ((size_t)threadIdx.x * 4 + t) * 128, 0, 1);
  }

  // ---- Phase 1: raw scores (Q fragments hoisted) ------------------------
  const v16bf qa0 = ld8x2_gap_bf(Qb + (size_t)(i0 + lm) * DKH + 0  + hi * 8);
  const v16bf qa1 = ld8x2_gap_bf(Qb + (size_t)(i0 + lm) * DKH + 32 + hi * 8);

  for (int jt = wave; jt < 32; jt += 4) {
    const int j0 = jt * 16;
    v16bf b0 = ld16_bf(Kb + (size_t)(j0 + lm) * DKH + 0  + hi * 8);
    v16bf b1 = ld16_bf(Kb + (size_t)(j0 + lm) * DKH + 32 + hi * 8);
    v8f acc = {};
    acc = wmma_bf16(qa0, b0, acc);
    acc = wmma_bf16(qa1, b1, acc);
    const int j = j0 + lm;
#pragma unroll
    for (int r = 0; r < 8; ++r) S[r + hi * 8][j] = acc[r] * 0.125f;
  }
  if (wave == 0) __builtin_amdgcn_s_wait_tensorcnt((short)0);  // TDM done
  __syncthreads();                                             // publish LDS

  // ---- Phase 2: bias + mask + softmax (8 threads per row) ---------------
  const int row = threadIdx.x >> 3;
  const int col = threadIdx.x & 7;
  float mx = -3.0e38f;
  for (int jj = col; jj < NSEQ; jj += 8) {
    float s = S[row][jj] + Sb[row][jj];
    if (Sm[row][jj] == 0) s = -1.0e9f;
    mx = fmaxf(mx, s);
  }
  red[row][col] = mx;
  __syncthreads();
#pragma unroll
  for (int t = 0; t < 8; ++t) mx = fmaxf(mx, red[row][t]);
  __syncthreads();
  float sum = 0.f;
  for (int jj = col; jj < NSEQ; jj += 8) {
    float s = S[row][jj] + Sb[row][jj];
    if (Sm[row][jj] == 0) s = -1.0e9f;
    const float ev = __expf(s - mx);
    S[row][jj] = ev;
    sum += ev;
  }
  red[row][col] = sum;
  __syncthreads();
  float tot = 0.f;
#pragma unroll
  for (int t = 0; t < 8; ++t) tot += red[row][t];
  const float inv = 1.0f / tot;
  for (int jj = col; jj < NSEQ; jj += 8) P[row][jj] = (bf16)(S[row][jj] * inv);
  __syncthreads();

  // ---- Phase 3: ctx = P @ V  (wave w owns d-cols w*16..w*16+15) ---------
  const int d0 = wave * 16;
  v8f acc = {};
  for (int j0 = 0; j0 < NSEQ; j0 += 32) {
    v16bf a  = ld8x2_gap_bf(&P[lm][j0 + hi * 8]);                       // LDS
    v16bf bm = ld16_bf(Vb + (size_t)(d0 + lm) * NSEQ + j0 + hi * 8);
    acc = wmma_bf16(a, bm, acc);
  }
  bf16* cp = ctx + ((size_t)b * NSEQ + i0) * D_MODEL + h * DKH + d0 + lm;
#pragma unroll
  for (int r = 0; r < 8; ++r) cp[(size_t)(r + hi * 8) * D_MODEL] = (bf16)acc[r];
}

// ---------------------------------------------------------------------------
// Kernel 3: output projection, 2x2 register-blocked.
// out(f32) = ctx(bf16) @ Wo^T + bo, Wo pre-converted bf16 (Wb slot 3).
// ---------------------------------------------------------------------------
__global__ __launch_bounds__(256) void out_proj_kernel(
    const bf16* __restrict__ ctx, const bf16* __restrict__ Wob,
    const float* __restrict__ bo, float* __restrict__ out)
{
  const int wave = threadIdx.x >> 5;
  const int lane = threadIdx.x & 31;
  const int lm   = lane & 15;
  const int hi   = lane >> 4;

  const int tile = blockIdx.x * 8 + wave;
  const int tm2  = tile >> 5;
  const int tn2  = tile & 31;
  const int r0 = tm2 * 32, c0 = tn2 * 32;

  const bf16* xrow0 = ctx + (size_t)(r0 + lm) * D_MODEL;
  const bf16* xrow1 = ctx + (size_t)(r0 + 16 + lm) * D_MODEL;
  const bf16* wrow0 = Wob + (size_t)(c0 + lm) * D_MODEL;
  const bf16* wrow1 = Wob + (size_t)(c0 + 16 + lm) * D_MODEL;

  v8f acc[2][2] = {};
#pragma unroll 2
  for (int k0 = 0; k0 < D_MODEL; k0 += 32) {
    v16bf a0 = ld8x2_gap_bf(xrow0 + k0 + hi * 8);
    v16bf a1 = ld8x2_gap_bf(xrow1 + k0 + hi * 8);
    v16bf b0 = ld16_bf(wrow0 + k0 + hi * 8);
    v16bf b1 = ld16_bf(wrow1 + k0 + hi * 8);
    acc[0][0] = wmma_bf16(a0, b0, acc[0][0]);
    acc[0][1] = wmma_bf16(a0, b1, acc[0][1]);
    acc[1][0] = wmma_bf16(a1, b0, acc[1][0]);
    acc[1][1] = wmma_bf16(a1, b1, acc[1][1]);
  }

#pragma unroll
  for (int ni = 0; ni < 2; ++ni) {
    const int o = c0 + ni * 16 + lm;
    const float bb = bo[o];
#pragma unroll
    for (int mi = 0; mi < 2; ++mi) {
#pragma unroll
      for (int r = 0; r < 8; ++r)
        out[(size_t)(r0 + mi * 16 + r + hi * 8) * D_MODEL + o] = acc[mi][ni][r] + bb;
    }
  }
}

// ---------------------------------------------------------------------------
// Host launcher.  Workspace layout (needs >= 72 MB):
//   [0,8M)    Wb  bf16 (Wq,Wk,Wv,Wo converted)
//   [8M,24M)  Qh  bf16   [24M,40M) Kh bf16   [40M,56M) Vt bf16
//   [56M,72M) ctx bf16
// ---------------------------------------------------------------------------
extern "C" void kernel_launch(void* const* d_in, const int* in_sizes, int n_in,
                              void* d_out, int out_size, void* d_ws, size_t ws_size,
                              hipStream_t stream) {
  (void)in_sizes; (void)n_in; (void)out_size; (void)ws_size;
  const float* q     = (const float*)d_in[0];
  const float* k     = (const float*)d_in[1];
  const float* v     = (const float*)d_in[2];
  const float* abias = (const float*)d_in[3];
  const int*   mask  = (const int*)  d_in[4];
  const float* Wq    = (const float*)d_in[5];
  const float* bq    = (const float*)d_in[6];
  const float* Wk    = (const float*)d_in[7];
  const float* bk    = (const float*)d_in[8];
  const float* Wv    = (const float*)d_in[9];
  const float* bv    = (const float*)d_in[10];
  const float* Wo    = (const float*)d_in[11];
  const float* bo    = (const float*)d_in[12];
  float* out = (float*)d_out;

  char* ws = (char*)d_ws;
  bf16* Wb  = (bf16*)(ws);
  bf16* Qh  = (bf16*)(ws + ((size_t)8  << 20));
  bf16* Kh  = (bf16*)(ws + ((size_t)24 << 20));
  bf16* Vt  = (bf16*)(ws + ((size_t)40 << 20));
  bf16* ctx = (bf16*)(ws + ((size_t)56 << 20));
  bf16* Wob = Wb + (size_t)3 * D_MODEL * D_MODEL;

  cvt_w_kernel<<<2048, 256, 0, stream>>>(Wq, Wk, Wv, Wo, Wb);
  qkv_proj_kernel<<<3072, 256, 0, stream>>>(q, k, v, Wb, bq, bk, bv, Qh, Kh, Vt);
  attn_kernel<<<NB * NH * (NSEQ / 16), 128, 0, stream>>>(Qh, Kh, Vt, abias, mask, ctx);
  out_proj_kernel<<<1024, 256, 0, stream>>>(ctx, Wob, bo, out);
}
